// ModalityQualityAwareEnhancement_64252710748728
// MI455X (gfx1250) — compile-verified, matrix operand order checked by reading, not run on hardware
//
#include <hip/hip_runtime.h>
#include <hip/hip_bf16.h>
#include <math.h>

typedef float  v8f  __attribute__((ext_vector_type(8)));
typedef __bf16 v16bf __attribute__((ext_vector_type(16)));

#define NVOX 262144ull     // 64^3 voxels per (m,c)
#define OUT_ENH (4ull*64ull*NVOX)

// ---- workspace layout (float offsets) ----
#define OFF_S     0ull                      // S[c][v] : 64 * 262144 floats (64 MB)
#define OFF_GAP   16777216ull               // gap sums [4][64]
#define OFF_DSUM  (OFF_GAP + 256ull)        // pooled sums [4][64][64bins]
#define OFF_AC    (OFF_DSUM + 16384ull)     // a_coarse [4][64]
#define OFF_SCAL  (OFF_AC + 256ull)         // qw[4], invT[4]
#define OFF_CSUM  (OFF_SCAL + 8ull)         // contrib sums [4][64] (atomic)
#define OFF_GATE  (OFF_CSUM + 256ull)       // gate [4][64]
#define OFF_GB    (OFF_GATE + 256ull)       // gate*bf [4][64]
#define OFF_WFA   (OFF_GB + 256ull)         // wf packed A-frags: 4096 bf16 (2048 floats)

__device__ __forceinline__ float sigmoidf_(float x){ return 1.0f/(1.0f+__expf(-x)); }

__device__ __forceinline__ unsigned short f2bf(float f){
  unsigned int u = __float_as_uint(f);
  u += 0x7FFFu + ((u >> 16) & 1u);        // round-to-nearest-even
  return (unsigned short)(u >> 16);
}

// per-axis trilinear params for 4 -> 64 resize (align_corners=False)
__device__ __forceinline__ void axisw(int i, int& lo, int& l1, float& fr){
  float cc = ((float)i + 0.5f) * 0.0625f - 0.5f;
  cc = fminf(fmaxf(cc, 0.0f), 3.0f);
  int l = (int)cc;
  fr = cc - (float)l;
  lo = l; l1 = min(l + 1, 3);
}

// ---------------- Kernel 1: spatial reductions (gap + 4x4x4 pooling) ----------------
__global__ void __launch_bounds__(256) k_reduce(const float* __restrict__ x,
                                                float* __restrict__ ws){
  __shared__ float part[256*4];
  __shared__ float tsum[256];
  int t = threadIdx.x;
  int mc = blockIdx.x;                       // m*64 + c
  // thread t owns 1024 contiguous voxels: fixed h = t>>2, w in [(t&3)*16, +16), all z
  const float* xp = x + (size_t)mc * NVOX + (size_t)t * 1024;
  float accz[4] = {0.f,0.f,0.f,0.f};
  for (int ww = 0; ww < 16; ++ww){
    #pragma unroll
    for (int zb = 0; zb < 4; ++zb){
      const float4* p = (const float4*)(xp + ww*64 + zb*16);
      float s = 0.f;
      #pragma unroll
      for (int j = 0; j < 4; ++j){ float4 q = p[j]; s += (q.x+q.y)+(q.z+q.w); }
      accz[zb] += s;
    }
  }
  #pragma unroll
  for (int zb = 0; zb < 4; ++zb) part[t*4 + zb] = accz[zb];
  tsum[t] = (accz[0]+accz[1])+(accz[2]+accz[3]);
  __syncthreads();
  if (t < 64){                               // bin = bx*16 + by*4 + bz
    int bx = t >> 4, by = (t >> 2) & 3, bz = t & 3;
    float s = 0.f;
    for (int hh = 0; hh < 16; ++hh) s += part[((bx*16+hh)*4 + by)*4 + bz];
    ws[OFF_DSUM + (size_t)mc*64 + t] = s;
  }
  for (int st = 128; st > 0; st >>= 1){
    __syncthreads();
    if (t < st) tsum[t] += tsum[t + st];
  }
  if (t == 0) ws[OFF_GAP + mc] = tsum[0];
}

// ---------------- Kernel 2: quality MLP + coarse attention MLP ----------------
__global__ void __launch_bounds__(256) k_small(float* __restrict__ ws,
    const float* __restrict__ wq1, const float* __restrict__ bq1,
    const float* __restrict__ wq2, const float* __restrict__ bq2,
    const float* __restrict__ wa1, const float* __restrict__ ba1,
    const float* __restrict__ wa2, const float* __restrict__ ba2,
    float* __restrict__ outScal){
  __shared__ float qws[4];
  int t = threadIdx.x;
  ws[OFF_CSUM + t] = 0.0f;                   // zero atomic accumulators (256)
  if (t < 4){
    int m = t;
    float hq[4] = {0.f,0.f,0.f,0.f};
    for (int c = 0; c < 64; ++c){
      float g = ws[OFF_GAP + m*64 + c] * (1.0f/262144.0f);
      #pragma unroll
      for (int k = 0; k < 4; ++k) hq[k] += wq1[k*64+c] * g;
    }
    #pragma unroll
    for (int k = 0; k < 4; ++k) hq[k] = fmaxf(hq[k] + bq1[k], 0.0f);
    float s0 = bq2[0], s1 = bq2[1];
    #pragma unroll
    for (int k = 0; k < 4; ++k){ s0 += wq2[k]*hq[k]; s1 += wq2[4+k]*hq[k]; }
    s0 = sigmoidf_(s0); s1 = sigmoidf_(s1);
    outScal[m]     = s0;                     // boundary
    outScal[4 + m] = s1;                     // semantic
    float q = 0.5f*(s0 + s1);
    qws[m] = q;
    ws[OFF_SCAL + m] = q;
  }
  {                                          // a_coarse: (m = t>>6, bin = t&63)
    int m = t >> 6, bin = t & 63;
    float hk[4] = {0.f,0.f,0.f,0.f};
    for (int c = 0; c < 64; ++c){
      float dv = ws[OFF_DSUM + ((size_t)(m*64 + c))*64 + bin] * (1.0f/4096.0f);
      #pragma unroll
      for (int k = 0; k < 4; ++k) hk[k] += wa1[k*64+c] * dv;
    }
    float a = ba2[0];
    #pragma unroll
    for (int k = 0; k < 4; ++k) a += wa2[k] * fmaxf(hk[k] + ba1[k], 0.0f);
    ws[OFF_AC + t] = sigmoidf_(a);
  }
  __syncthreads();
  if (t < 4){
    float T = qws[0]+qws[1]+qws[2]+qws[3];
    ws[OFF_SCAL + 4 + t] = 1.0f/(T - qws[t]);
  }
}

// ---------------- Kernel 3: S = sum_m x*a*qw, plus per-(m,c) contrib sums ----------------
__global__ void __launch_bounds__(256) k_contrib(const float* __restrict__ x,
                                                 float* __restrict__ ws){
  __shared__ float aC[256];
  __shared__ float qwl[4];
  __shared__ float red[256*4];
  int t = threadIdx.x;
  aC[t] = ws[OFF_AC + t];
  if (t < 4) qwl[t] = ws[OFF_SCAL + t];
  __syncthreads();
  int c = blockIdx.x & 63;
  int chunk = blockIdx.x >> 6;
  int v0 = chunk*4096 + t*16;                // 16 contiguous voxels: h,w fixed
  int h = v0 >> 12, w = (v0 >> 6) & 63, z0 = v0 & 63;
  int lh,lh1,lw,lw1; float fh,fw;
  axisw(h, lh, lh1, fh);
  axisw(w, lw, lw1, fw);
  float bl[4][4];                            // bilinear(h,w) per m per lz
  #pragma unroll
  for (int m = 0; m < 4; ++m){
    const float* am = aC + m*64;
    #pragma unroll
    for (int lz = 0; lz < 4; ++lz){
      float a00 = am[lh*16  + lw*4  + lz];
      float a01 = am[lh*16  + lw1*4 + lz];
      float a10 = am[lh1*16 + lw*4  + lz];
      float a11 = am[lh1*16 + lw1*4 + lz];
      bl[m][lz] = (a00*(1.f-fw)+a01*fw)*(1.f-fh) + (a10*(1.f-fw)+a11*fw)*fh;
    }
  }
  float cs[4] = {0.f,0.f,0.f,0.f};
  float* Sp = ws + OFF_S + (size_t)c * NVOX;
  #pragma unroll
  for (int g = 0; g < 4; ++g){
    int vv = v0 + g*4;
    float aq[4][4];                          // [j][m]
    #pragma unroll
    for (int j = 0; j < 4; ++j){
      int lz, lz1; float fz;
      axisw(z0 + g*4 + j, lz, lz1, fz);
      #pragma unroll
      for (int m = 0; m < 4; ++m)
        aq[j][m] = (bl[m][lz]*(1.f-fz) + bl[m][lz1]*fz) * qwl[m];
    }
    float4 sacc = {0.f,0.f,0.f,0.f};
    #pragma unroll
    for (int m = 0; m < 4; ++m){
      const float* xp = x + ((size_t)(m*64 + c))*NVOX + vv;
      __builtin_prefetch(xp + 4096, 0, 1);   // warm L2 for next chunk
      float4 xv = *(const float4*)xp;
      float c0 = xv.x*aq[0][m], c1 = xv.y*aq[1][m];
      float c2 = xv.z*aq[2][m], c3 = xv.w*aq[3][m];
      sacc.x += c0; sacc.y += c1; sacc.z += c2; sacc.w += c3;
      cs[m] += (c0+c1)+(c2+c3);
    }
    *(float4*)(Sp + vv) = sacc;
  }
  #pragma unroll
  for (int m = 0; m < 4; ++m) red[t*4 + m] = cs[m];
  for (int st = 128; st > 0; st >>= 1){
    __syncthreads();
    if (t < st){
      #pragma unroll
      for (int m = 0; m < 4; ++m) red[t*4+m] += red[(t+st)*4+m];
    }
  }
  __syncthreads();
  if (t < 4) atomicAdd(&ws[OFF_CSUM + t*64 + c], red[t]);
}

// ---------------- Kernel 4: gate MLP + pack wf into WMMA A-fragments ----------------
__global__ void __launch_bounds__(256) k_gate(float* __restrict__ ws,
    const float* __restrict__ wf,  const float* __restrict__ bf,
    const float* __restrict__ wg1, const float* __restrict__ bg1,
    const float* __restrict__ wg2, const float* __restrict__ bg2){
  __shared__ float me[256];
  __shared__ float gg[256];
  __shared__ float hks[16];
  int t = threadIdx.x;
  int m = t >> 6, d = t & 63;
  {
    int c = d;
    float s = 0.f;
    #pragma unroll
    for (int mm = 0; mm < 4; ++mm) s += ws[OFF_CSUM + mm*64 + c];
    float own = ws[OFF_CSUM + m*64 + c];
    me[t] = (s - own) * (ws[OFF_SCAL + 4 + m] * (1.0f/262144.0f));   // meanEnh[m][c]
  }
  __syncthreads();
  {
    float acc = bf[d];
    for (int c = 0; c < 64; ++c) acc += wf[d*64 + c] * me[m*64 + c];
    gg[t] = acc;                                                     // g[m][d]
  }
  __syncthreads();
  if (t < 16){
    int m2 = t >> 2, k = t & 3;
    float hh = bg1[k];
    for (int c = 0; c < 64; ++c) hh += wg1[k*64 + c] * gg[m2*64 + c];
    hks[t] = fmaxf(hh, 0.f);
  }
  __syncthreads();
  {
    float ga = bg2[d];
    #pragma unroll
    for (int k = 0; k < 4; ++k) ga += wg2[d*4 + k] * hks[m*4 + k];
    ga = sigmoidf_(ga);
    ws[OFF_GATE + t] = ga;
    ws[OFF_GB   + t] = ga * bf[d];
  }
  // pack wf (64x64) into 8 bf16 A-fragments (rt in 0..3, kt in 0..1) for 16x16x32 WMMA
  unsigned short* wfa = (unsigned short*)(ws + OFF_WFA);
  int f = t >> 5, l = t & 31;
  int rt = f >> 1, kt = f & 1;
  #pragma unroll
  for (int i = 0; i < 16; ++i){
    int row = rt*16 + (l & 15);
    int col = kt*32 + ((i & 8) ? 16 : 0) + (l >> 4)*8 + (i & 7);
    wfa[(f*32 + l)*16 + i] = f2bf(wf[row*64 + col]);
  }
}

// ---------------- Kernel 5: fused enh -> WMMA matmul -> gated residual ----------------
// Block index is chunk-major: the 4 blocks (one per modality) sharing one S chunk are
// adjacent in launch order so the S tile is served from L2 after the first reader.
__global__ void __launch_bounds__(256, 1) k_fuse(const float* __restrict__ x,
                                                 float* __restrict__ ws,
                                                 float* __restrict__ out){
  __shared__ float xT[64*128];               // x tile [c][v]; reused as output stage (32 KB)
  __shared__ unsigned short enhB[128*64];    // enh bf16 tile [v][c] (16 KB)
  __shared__ float aqv[128];
  __shared__ float gateL[64], gbL[64], aCm[64];
  int t = threadIdx.x;
  int m  = blockIdx.x & 3;
  int vb = (blockIdx.x >> 2) * 128;
  if (t < 64){
    gateL[t] = ws[OFF_GATE + m*64 + t];
    gbL[t]   = ws[OFF_GB   + m*64 + t];
    aCm[t]   = ws[OFF_AC   + m*64 + t];
  }
  const float* xm = x + (size_t)m * 64ull * NVOX;
  #pragma unroll
  for (int i = 0; i < 8; ++i){               // coalesced float4 tile load of x
    int f = t + i*256;
    int c = f >> 5;
    int voff = (f & 31) * 4;
    const float* xp = xm + (size_t)c*NVOX + vb + voff;
    __builtin_prefetch(xp + 128, 0, 1);      // warm L2 for next chunk
    float4 q = *(const float4*)xp;
    *(float4*)(xT + c*128 + voff) = q;
  }
  float qw_m = ws[OFF_SCAL + m];
  float invT = ws[OFF_SCAL + 4 + m];
  __syncthreads();
  if (t < 128){                              // trilinear a(v) * qw
    int v = vb + t;
    int h = v >> 12, w = (v >> 6) & 63, z = v & 63;
    int lh,lh1,lw,lw1,lz,lz1; float fh,fw,fz;
    axisw(h, lh, lh1, fh); axisw(w, lw, lw1, fw); axisw(z, lz, lz1, fz);
    float a00 = aCm[lh*16 +lw*4 +lz]*(1.f-fz) + aCm[lh*16 +lw*4 +lz1]*fz;
    float a01 = aCm[lh*16 +lw1*4+lz]*(1.f-fz) + aCm[lh*16 +lw1*4+lz1]*fz;
    float a10 = aCm[lh1*16+lw*4 +lz]*(1.f-fz) + aCm[lh1*16+lw*4 +lz1]*fz;
    float a11 = aCm[lh1*16+lw1*4+lz]*(1.f-fz) + aCm[lh1*16+lw1*4+lz1]*fz;
    float a = (a00*(1.f-fw)+a01*fw)*(1.f-fh) + (a10*(1.f-fw)+a11*fw)*fh;
    aqv[t] = a * qw_m;
  }
  __syncthreads();
  const float* Sb = ws + OFF_S;
  #pragma unroll
  for (int i = 0; i < 8; ++i){               // stream S, form enh in bf16 [v][c]
    int f = t + i*256;
    int c = f >> 5;
    int voff = (f & 31) * 4;
    float4 s4 = *(const float4*)(Sb + (size_t)c*NVOX + vb + voff);
    float4 x4 = *(const float4*)(xT + c*128 + voff);
    enhB[(voff+0)*64 + c] = f2bf((s4.x - x4.x*aqv[voff+0]) * invT);
    enhB[(voff+1)*64 + c] = f2bf((s4.y - x4.y*aqv[voff+1]) * invT);
    enhB[(voff+2)*64 + c] = f2bf((s4.z - x4.z*aqv[voff+2]) * invT);
    enhB[(voff+3)*64 + c] = f2bf((s4.w - x4.w*aqv[voff+3]) * invT);
  }
  __syncthreads();
  // ---- per-wave WMMA: 64 (out ch) x 16 (voxels) tile via 8x v_wmma_f32_16x16x32_bf16
  int lane = t & 31, wv = t >> 5;
  int vloc = wv*16 + (lane & 15);
  int khalf = (lane >> 4) * 8;
  union VB { v16bf v; unsigned int u[8]; };
  VB af[8];
  const unsigned int* wfa = (const unsigned int*)(ws + OFF_WFA);
  #pragma unroll
  for (int f = 0; f < 8; ++f)
    #pragma unroll
    for (int p = 0; p < 8; ++p)
      af[f].u[p] = wfa[(f*32 + lane)*8 + p];
  VB b0, b1;
  #pragma unroll
  for (int p = 0; p < 8; ++p){
    int K = ((p & 4) ? 16 : 0) + khalf + (p & 3)*2;   // even -> 32b aligned pair K,K+1
    b0.u[p] = *(const unsigned int*)&enhB[vloc*64 + K];
    b1.u[p] = *(const unsigned int*)&enhB[vloc*64 + 32 + K];
  }
  v8f acc[4];
  #pragma unroll
  for (int rt = 0; rt < 4; ++rt){
    v8f cz = {0.f,0.f,0.f,0.f,0.f,0.f,0.f,0.f};
    cz = __builtin_amdgcn_wmma_f32_16x16x32_bf16(false, af[rt*2+0].v, false, b0.v,
                                                 (short)0, cz, false, false);
    cz = __builtin_amdgcn_wmma_f32_16x16x32_bf16(false, af[rt*2+1].v, false, b1.v,
                                                 (short)0, cz, false, false);
    acc[rt] = cz;
  }
  // epilogue: finish out = x + gate*mm + gate*bf in place in the xT tile.
  // Each (d, vloc) element is owned by exactly one lane -> safe in-place overwrite.
  int rowoff = (lane >> 4) * 8;
  #pragma unroll
  for (int rt = 0; rt < 4; ++rt){
    #pragma unroll
    for (int r = 0; r < 8; ++r){
      int d = rt*16 + rowoff + r;
      float fv = acc[rt][r];
      xT[d*128 + vloc] = xT[d*128 + vloc] + gateL[d]*fv + gbL[d];
    }
  }
  __syncthreads();
  float* om = out + (size_t)m * 64ull * NVOX;
  #pragma unroll
  for (int i = 0; i < 8; ++i){               // coalesced b128 stores of the staged tile
    int f = t + i*256;
    int c = f >> 5;
    int voff = (f & 31) * 4;
    float4 q = *(const float4*)(xT + c*128 + voff);
    *(float4*)(om + (size_t)c*NVOX + vb + voff) = q;
  }
}

extern "C" void kernel_launch(void* const* d_in, const int* in_sizes, int n_in,
                              void* d_out, int out_size, void* d_ws, size_t ws_size,
                              hipStream_t stream){
  const float* x   = (const float*)d_in[0];
  const float* wq1 = (const float*)d_in[1];
  const float* bq1 = (const float*)d_in[2];
  const float* wq2 = (const float*)d_in[3];
  const float* bq2 = (const float*)d_in[4];
  const float* wa1 = (const float*)d_in[5];
  const float* ba1 = (const float*)d_in[6];
  const float* wa2 = (const float*)d_in[7];
  const float* ba2 = (const float*)d_in[8];
  const float* wf  = (const float*)d_in[9];
  const float* bf  = (const float*)d_in[10];
  const float* wg1 = (const float*)d_in[11];
  const float* bg1 = (const float*)d_in[12];
  const float* wg2 = (const float*)d_in[13];
  const float* bg2 = (const float*)d_in[14];
  float* out = (float*)d_out;
  float* ws  = (float*)d_ws;

  k_reduce <<<256,  256, 0, stream>>>(x, ws);
  k_small  <<<1,    256, 0, stream>>>(ws, wq1,bq1,wq2,bq2, wa1,ba1,wa2,ba2, out + OUT_ENH);
  k_contrib<<<4096, 256, 0, stream>>>(x, ws);
  k_gate   <<<1,    256, 0, stream>>>(ws, wf, bf, wg1, bg1, wg2, bg2);
  k_fuse   <<<8192, 256, 0, stream>>>(x, ws, out);
}
